// VectorQuantizer_7395933684078
// MI455X (gfx1250) — matrix operand order repaired and checked by reading, not compiled, not run on hardware
//
#include <hip/hip_runtime.h>
#include <hip/hip_bf16.h>
#include <stdint.h>

typedef __attribute__((ext_vector_type(16))) _Float16 v16h;
typedef __attribute__((ext_vector_type(8)))  float    v8f;

#define N_ROWS   16384      // 16 * 32 * 32
#define N_E      8192
#define E_DIM    64
#define NCH      128        // chunks of 64 codes
#define EPSV     1e-5f
#define NEG_BIG  (-1.0e30f)
#define LN2F     0.6931471805599453f
#define K2F      144.26950408889634f     // 100 * log2(e)
#define K4F      288.53900817779268f     // 200 * log2(e)

// workspace layout (bytes):
//   [0, 1048576)            : f16 codebook copy (8192 x 64)
//   then floats: [0]=ent sum, [1]=loss sum, [2..8193]=avg_prob,
//                [8194..16385]=K2F*cnorm
#define WS_F_OFF  (N_E * E_DIM / 2)      // float offset of scalar region

// ---------------------------------------------------------------------------
// Kernel 0: zero the accumulator region (ent sum, loss sum, avg_prob[8192])
// ---------------------------------------------------------------------------
__global__ __launch_bounds__(256) void vq_zero(float* __restrict__ wsF) {
  int i = blockIdx.x * 256 + threadIdx.x;
  if (i < 2 + N_E) wsF[i] = 0.0f;
}

// ---------------------------------------------------------------------------
// Kernel 1: pre-scaled code norms  cnS[k] = 100*log2e * ||emb_k||^2
// ---------------------------------------------------------------------------
__global__ __launch_bounds__(256) void vq_cnorm(const float* __restrict__ emb,
                                                float* __restrict__ cnS) {
  int k = blockIdx.x * 256 + threadIdx.x;          // 8192 threads
  const float* e = emb + (size_t)k * E_DIM;
  float s = 0.0f;
#pragma unroll
  for (int d = 0; d < E_DIM; ++d) s = fmaf(e[d], e[d], s);
  cnS[k] = K2F * s;
}

// ---------------------------------------------------------------------------
// Kernel 2: one-time f32 -> f16 codebook conversion (feeds async LDS copies)
// ---------------------------------------------------------------------------
__global__ __launch_bounds__(256) void vq_half(const float* __restrict__ emb,
                                               _Float16* __restrict__ embH) {
  int i = blockIdx.x * 256 + threadIdx.x;          // 524288 elements
  embH[i] = (_Float16)emb[i];
}

// ---------------------------------------------------------------------------
// Kernel 3: fused GEMM (WMMA f16 -> f32) + online softmax / argmax + avg_prob
//   logits (log2 domain) lb[n,k] = 100*log2e * (2 * z_n . e_k - ||e_k||^2)
//   (row-constant dropped: softmax and argmin are invariant to it).
//   Block: 256 threads = 8 wave32; each wave owns a 16-row tile, FOUR 16x16
//   accumulator tiles per chunk of 64 codes (8 WMMAs/chunk). Chunks are
//   TRIPLE-BUFFERED via global_load_async_to_lds_b128 (ASYNCcnt<=2 keeps one
//   chunk in flight under the compute). Hot loop keeps PER-LANE online state
//   (no cross-lane ops, raw v_exp_f32 transcendentals, one rescale per chunk
//   per row); lanes merge branchlessly once at the end of pass A.
// ---------------------------------------------------------------------------
__global__ __launch_bounds__(256) void vq_main(const float* __restrict__ z,
                                               const _Float16* __restrict__ embH,
                                               const float* __restrict__ cnS,
                                               float* __restrict__ outIdxF,
                                               float* __restrict__ gAvg,
                                               float* __restrict__ gEnt) {
  __shared__ _Float16 sB[3][64 * 72];  // 64 codes x 64 dims, padded stride 72
  __shared__ float    sAvg[N_E];       // block-local avg_prob accumulator

  const int t = threadIdx.x;
  for (int k = t; k < N_E; k += 256) sAvg[k] = 0.0f;

  const int lane    = t & 31;
  const int wave    = t >> 5;
  const int rowBase = blockIdx.x * 128 + wave * 16;
  const int r       = lane & 15;       // row-in-tile (A) / code-in-tile (B)
  const int half    = lane >> 4;       // 0 or 1
  const int o       = half * 8;        // unused; A frags also use kb
  const int kb      = half * 16;       // B-frag K offset

  // ---- async staging: each thread moves two 16B packets per chunk --------
  const int scode = t >> 2;            // 0..63
  const int sdim  = (t & 3) * 16;      // 0,16,32,48
  const uint32_t ldsB0  = (uint32_t)(uintptr_t)&sB[0][scode * 72 + sdim];
  const uint64_t gB     = (uint64_t)(uintptr_t)embH;
  const uint32_t gOffB0 = (uint32_t)((scode * E_DIM + sdim) * 2);

  auto issue = [&](int ch) {           // async-copy chunk ch into buffer ch%3
    const uint32_t lds = ldsB0 + (uint32_t)(ch % 3) * (64u * 72u * 2u);
    const uint32_t go  = gOffB0 + (uint32_t)ch * 8192u;   // 64 codes * 128 B
    asm volatile("global_load_async_to_lds_b128 %0, %1, %2"
                 :: "v"(lds), "v"(go), "s"(gB) : "memory");
    asm volatile("global_load_async_to_lds_b128 %0, %1, %2 offset:16"
                 :: "v"(lds), "v"(go), "s"(gB) : "memory");
  };
  auto waitBar = [&](int ch) {         // chunk ch resident; ch+1 may stay in flight
    if (ch + 1 < NCH) asm volatile("s_wait_asynccnt 0x2" ::: "memory");
    else              asm volatile("s_wait_asynccnt 0x0" ::: "memory");
    __syncthreads();
  };

  // ---- A fragments: rows rowBase..rowBase+15, dims 0..63 (f32 -> f16) ----
  // z layout is [B=16, C=64, H=32, W=32]; zf[n][c] = z[(b*64+c)*1024 + hw]
  const int   nrow = rowBase + r;
  const int   bimg = nrow >> 10;
  const int   hw   = nrow & 1023;
  const float* zr  = z + (size_t)bimg * 64 * 1024 + hw;

  v16h a0, a1;
#pragma unroll
  for (int j = 0; j < 16; ++j) {
    int K = kb + j;                                  // contiguous K order (matches B)
    a0[j] = (_Float16)zr[(size_t)K * 1024];          // dims 0..31
    a1[j] = (_Float16)zr[(size_t)(K + 32) * 1024];   // dims 32..63
  }

  // compute the four 16x16 accumulator tiles from buffer ch%3
  auto mma4 = [&](int ch, v8f* acc) {
    const _Float16* B = &sB[ch % 3][0];
#pragma unroll
    for (int tile = 0; tile < 4; ++tile) {
      const int row = (tile * 16 + r) * 72;
      v16h blo, bhi;
#pragma unroll
      for (int j = 0; j < 16; ++j) {                 // contiguous K order (matches A)
        blo[j] = B[row + kb + j];
        bhi[j] = B[row + 32 + kb + j];
      }
      acc[tile] = __builtin_amdgcn_wmma_f32_16x16x32_f16(
          false, a0, false, blo, (short)0, acc[tile], false, false);
      acc[tile] = __builtin_amdgcn_wmma_f32_16x16x32_f16(
          false, a1, false, bhi, (short)0, acc[tile], false, false);
    }
  };

  // Per-lane partial online state (log2 domain) over this lane's columns.
  float runm[8], runs[8], bestv[8];
  int   besti[8];
#pragma unroll
  for (int i = 0; i < 8; ++i) {
    runm[i] = NEG_BIG; runs[i] = 0.0f; bestv[i] = NEG_BIG; besti[i] = 0;
  }

  // =========================== PASS A =====================================
  issue(0); issue(1);
  for (int ch = 0; ch < NCH; ++ch) {
    waitBar(ch);
    if (ch + 2 < NCH) issue(ch + 2);
    v8f acc[4] = {};
    mma4(ch, acc);
    float ccs[4];
#pragma unroll
    for (int tile = 0; tile < 4; ++tile)
      ccs[tile] = cnS[ch * 64 + tile * 16 + r];      // pre-scaled by K2F
#pragma unroll
    for (int i = 0; i < 8; ++i) {
      float lb[4];
#pragma unroll
      for (int tile = 0; tile < 4; ++tile) {
        lb[tile] = fmaf(K4F, acc[tile][i], -ccs[tile]);
        bool tk  = lb[tile] > bestv[i];              // strict > keeps earliest
        bestv[i] = tk ? lb[tile] : bestv[i];
        besti[i] = tk ? (ch * 64 + tile * 16 + r) : besti[i];
      }
      float tm = fmaxf(fmaxf(lb[0], lb[1]), fmaxf(lb[2], lb[3]));
      float nm = fmaxf(runm[i], tm);
      float es = (__builtin_amdgcn_exp2f(lb[0] - nm) +
                  __builtin_amdgcn_exp2f(lb[1] - nm)) +
                 (__builtin_amdgcn_exp2f(lb[2] - nm) +
                  __builtin_amdgcn_exp2f(lb[3] - nm));
      runs[i] = fmaf(runs[i], __builtin_amdgcn_exp2f(runm[i] - nm), es);
      runm[i] = nm;
    }
  }

  // ---- one-time branchless merge across the 16 lanes of each half --------
#pragma unroll
  for (int i = 0; i < 8; ++i) {
    float v = bestv[i]; int vi = besti[i];
    float m_ = runm[i]; float s_ = runs[i];
#pragma unroll
    for (int msk = 1; msk <= 8; msk <<= 1) {
      float ov = __shfl_xor(v, msk, 32);
      int   oi = __shfl_xor(vi, msk, 32);
      bool  take = (ov > v) || ((ov == v) && (oi < vi));
      v  = take ? ov : v;
      vi = take ? oi : vi;
      float om = __shfl_xor(m_, msk, 32);
      float os = __shfl_xor(s_, msk, 32);
      float nm = fmaxf(m_, om);
      s_ = fmaf(s_, __builtin_amdgcn_exp2f(m_ - nm),
                os * __builtin_amdgcn_exp2f(om - nm));
      m_ = nm;
    }
    bestv[i] = v; besti[i] = vi; runm[i] = m_; runs[i] = s_;
  }

  // ---- write argmin indices (rows i and i+8 from the two halves) ---------
  if (r == 0) {
    int rb = rowBase + half * 8;
#pragma unroll
    for (int i = 0; i < 8; ++i) outIdxF[rb + i] = (float)besti[i];
  }

  float invs[8];
#pragma unroll
  for (int i = 0; i < 8; ++i) invs[i] = __builtin_amdgcn_rcpf(runs[i]);

  // =========================== PASS B =====================================
  float entAcc = 0.0f;                  // in log2 units; * ln2 at the end
  issue(0); issue(1);
  for (int ch = 0; ch < NCH; ++ch) {
    waitBar(ch);
    if (ch + 2 < NCH) issue(ch + 2);
    v8f acc[4] = {};
    mma4(ch, acc);
    float ccs[4], psum[4];
#pragma unroll
    for (int tile = 0; tile < 4; ++tile) {
      ccs[tile]  = cnS[ch * 64 + tile * 16 + r];
      psum[tile] = 0.0f;
    }
#pragma unroll
    for (int i = 0; i < 8; ++i) {
#pragma unroll
      for (int tile = 0; tile < 4; ++tile) {
        float lb = fmaf(K4F, acc[tile][i], -ccs[tile]);
        float p  = __builtin_amdgcn_exp2f(lb - runm[i]) * invs[i];
        entAcc  -= p * __builtin_amdgcn_logf(fmaxf(p, EPSV));  // clamped
        psum[tile] += p;
      }
    }
    // each column is owned by two lanes (rows 0-7 and 8-15): both add.
#pragma unroll
    for (int tile = 0; tile < 4; ++tile)
      atomicAdd(&sAvg[ch * 64 + tile * 16 + r], psum[tile]);   // ds_add_f32
  }

  // per-sample entropy: wave covers all 16 rows x all columns of its tile
#pragma unroll
  for (int m = 1; m <= 16; m <<= 1) entAcc += __shfl_xor(entAcc, m, 32);
  if (lane == 0) atomicAdd(gEnt, entAcc * LN2F);

  __syncthreads();
  for (int k = t; k < N_E; k += 256) atomicAdd(&gAvg[k], sAvg[k]);
}

// ---------------------------------------------------------------------------
// Kernel 4: gather quantized output (NCHW) + commitment-loss accumulation
//   loss = (1 + 0.25) * mean((z_q - z)^2)    (stop_gradient is identity fwd)
// ---------------------------------------------------------------------------
__global__ __launch_bounds__(256) void vq_quant(const float* __restrict__ z,
                                                const float* __restrict__ emb,
                                                const float* __restrict__ idxF,
                                                float* __restrict__ outQ,
                                                float* __restrict__ gLoss) {
  int g  = blockIdx.x * 256 + threadIdx.x;     // index into [B,C,H,W]
  int bc = g >> 10, hw = g & 1023;
  int b  = bc >> 6, c = bc & 63;
  int nrow = (b << 10) + hw;
  int idx  = (int)idxF[nrow];
  float e  = emb[(size_t)idx * E_DIM + c];
  outQ[g]  = e;                                 // straight-through fwd value
  float d  = e - z[g];
  float s  = d * d;
#pragma unroll
  for (int m = 1; m <= 16; m <<= 1) s += __shfl_xor(s, m, 32);
  if ((threadIdx.x & 31) == 0) atomicAdd(gLoss, s);
}

// ---------------------------------------------------------------------------
// Kernel 5: codebook entropy of avg_prob + final scalars
// ---------------------------------------------------------------------------
__global__ __launch_bounds__(256) void vq_final(const float* __restrict__ gAvg,
                                                const float* __restrict__ gEnt,
                                                const float* __restrict__ gLoss,
                                                float* __restrict__ outLoss,
                                                float* __restrict__ outEnt) {
  __shared__ float red[8];
  int t = threadIdx.x;
  float acc = 0.0f;                    // log2 units
  for (int k = t; k < N_E; k += 256) {
    float a = gAvg[k] * (1.0f / (float)N_ROWS);
    acc -= a * __builtin_amdgcn_logf(fmaxf(a, EPSV));
  }
#pragma unroll
  for (int m = 1; m <= 16; m <<= 1) acc += __shfl_xor(acc, m, 32);
  if ((t & 31) == 0) red[t >> 5] = acc;
  __syncthreads();
  if (t == 0) {
    float cb = 0.0f;
#pragma unroll
    for (int w = 0; w < 8; ++w) cb += red[w];
    cb *= LN2F;
    outEnt[0]  = gEnt[0] * (1.0f / (float)N_ROWS) - cb;  // per_sample - gamma*codebook
    outLoss[0] = 1.25f * gLoss[0] * (1.0f / ((float)N_ROWS * (float)E_DIM));
  }
}

// ---------------------------------------------------------------------------
extern "C" void kernel_launch(void* const* d_in, const int* in_sizes, int n_in,
                              void* d_out, int out_size, void* d_ws, size_t ws_size,
                              hipStream_t stream) {
  (void)in_sizes; (void)n_in; (void)out_size; (void)ws_size;
  const float* z   = (const float*)d_in[0];   // [16,64,32,32]
  const float* emb = (const float*)d_in[1];   // [8192,64]

  float* out     = (float*)d_out;
  float* outQ    = out;                 // 65536 quantized (NCHW)
  float* outLoss = out + 65536;         // scalar
  float* outEnt  = out + 65537;         // scalar
  float* outIdx  = out + 65538;         // 16384 indices (as float)

  _Float16* embH = (_Float16*)d_ws;     // 16B-aligned f16 codebook copy
  float* wsF   = (float*)d_ws + WS_F_OFF;
  float* gEnt  = wsF + 0;               // per-sample entropy sum
  float* gLoss = wsF + 1;               // sum of squared diffs
  float* gAvg  = wsF + 2;               // avg_prob accumulator [8192]
  float* cnS   = wsF + 2 + N_E;         // K2F * ||e_k||^2 [8192]

  vq_zero <<<(2 + N_E + 255) / 256, 256, 0, stream>>>(wsF);
  vq_cnorm<<<N_E / 256,             256, 0, stream>>>(emb, cnS);
  vq_half <<<(N_E * E_DIM) / 256,   256, 0, stream>>>(emb, embH);
  vq_main <<<N_ROWS / 128,          256, 0, stream>>>(z, embH, cnS, outIdx, gAvg, gEnt);
  vq_quant<<<(N_ROWS * E_DIM) / 256, 256, 0, stream>>>(z, emb, outIdx, outQ, gLoss);
  vq_final<<<1,                     256, 0, stream>>>(gAvg, gEnt, gLoss, outLoss, outEnt);
}
